// NT_Xent_45397804319452
// MI455X (gfx1250) — matrix operand order repaired and checked by reading, not compiled, not run on hardware
//
#include <hip/hip_runtime.h>
#include <math.h>

typedef __attribute__((ext_vector_type(16))) _Float16 v16h;
typedef __attribute__((ext_vector_type(8)))  _Float16 v8h;
typedef __attribute__((ext_vector_type(8)))  float    v8f;
typedef __attribute__((ext_vector_type(4)))  float    v4f;

#define NROW   4096
#define DDIM   256
#define TWO_N  8192
#define INV_T  2.0f      // 1/temperature
#define MAXS   2.0f      // upper bound on masked sim values (cos<=1, /T)
#define EPSN   1e-8f
#define INV_N  (1.0f/4096.0f)

#define NB     128       // columns staged per batch
#define NBATCH (TWO_N / NB)   // 64

// ---------------------------------------------------------------- zero output
__global__ void zero_out(float* out) {
    if (threadIdx.x < 2) out[threadIdx.x] = 0.0f;
}

// ------------------------------------------------- row-normalize, f32 -> f16
__global__ __launch_bounds__(256) void normalize_f16(
    const float* __restrict__ zi, const float* __restrict__ zj,
    _Float16* __restrict__ H)
{
    int wave = threadIdx.x >> 5;
    int lane = threadIdx.x & 31;
    int row  = blockIdx.x * 8 + wave;
    const float* src = (row < NROW) ? (zi + (size_t)row * DDIM)
                                    : (zj + (size_t)(row - NROW) * DDIM);
    v4f x0 = *(const v4f*)(src + lane * 8);
    v4f x1 = *(const v4f*)(src + lane * 8 + 4);
    float ss = 0.0f;
#pragma unroll
    for (int j = 0; j < 4; ++j) ss += x0[j]*x0[j] + x1[j]*x1[j];
#pragma unroll
    for (int m = 1; m < 32; m <<= 1) ss += __shfl_xor(ss, m, 32);
    float inv = 1.0f / fmaxf(sqrtf(ss), EPSN);
    v8h o;
#pragma unroll
    for (int j = 0; j < 4; ++j) {
        o[j]     = (_Float16)(x0[j] * inv);
        o[4 + j] = (_Float16)(x1[j] * inv);
    }
    *(v8h*)(H + (size_t)row * DDIM + lane * 8) = o;
}

// --------------------------------------------- positives: pos_i[r]==pos_j[r]
__global__ __launch_bounds__(256) void pos_subtract(
    const _Float16* __restrict__ H, float* __restrict__ out)
{
    int wave = threadIdx.x >> 5;
    int lane = threadIdx.x & 31;
    int r    = blockIdx.x * 8 + wave;
    v8h va = *(const v8h*)(H + (size_t)r          * DDIM + lane * 8);
    v8h vb = *(const v8h*)(H + (size_t)(r + NROW) * DDIM + lane * 8);
    float d = 0.0f;
#pragma unroll
    for (int j = 0; j < 8; ++j) d += (float)va[j] * (float)vb[j];
#pragma unroll
    for (int m = 1; m < 32; m <<= 1) d += __shfl_xor(d, m, 32);
    if (lane == 0) {
        float p = -(d * INV_T) * INV_N;
        atomicAdd(&out[0], p);
        atomicAdd(&out[1], p);
    }
}

// --------------------------------------------------------- main WMMA kernel
// block = 256 threads = 8 waves, owns 32 rows (2 row-tiles of 16).
// Column panel of 128 columns staged in LDS via async global->LDS loads,
// double-buffered (ASYNCcnt). Wave w: row tile w&1, column tiles
// {w>>1, (w>>1)+4} of the staged panel. A fragments live in registers.
__global__ __launch_bounds__(256) void ntxent_gram(
    const _Float16* __restrict__ H, float* __restrict__ out)
{
    __shared__ __attribute__((aligned(16))) _Float16 buf[2][NB * DDIM]; // 2x64KB
    __shared__ float rowsum[32];
    int tid = threadIdx.x;
    if (tid < 32) rowsum[tid] = 0.0f;

    int wave = tid >> 5;
    int lane = tid & 31;
    int l15  = lane & 15;
    int hsel = lane >> 4;                 // 0 or 1 (half-wave select)
    int rt   = wave & 1;                  // row tile within block
    int cb   = wave >> 1;                 // column sub-tile group (0..3)
    int m0   = blockIdx.x * 32 + rt * 16;

    // ---- A fragments: 16x32 f16 per k-step (ISA layout), loaded once
    v16h a[8];
    {
        const _Float16* arow = H + (size_t)(m0 + l15) * DDIM + hsel * 8;
#pragma unroll
        for (int ks = 0; ks < 8; ++ks) {
            v8h lo = *(const v8h*)(arow + ks * 32);
            v8h hi = *(const v8h*)(arow + ks * 32 + 16);
            a[ks] = __builtin_shufflevector(lo, hi,
                0,1,2,3,4,5,6,7,8,9,10,11,12,13,14,15);
        }
    }

    // ---- async stage one 128-column panel (64KB) into buf[sel]
    // thread t moves 16 chunks of 16B: chunk c = t + i*256
    auto stage = [&](int b, int sel) {
        const char* src = (const char*)(H + (size_t)b * NB * DDIM);
        unsigned dstbase = (unsigned)(uintptr_t)(&buf[sel][0]);
#pragma unroll
        for (int i = 0; i < 16; ++i) {
            unsigned off = (unsigned)(tid + i * 256) * 16u;
            unsigned long long ga = (unsigned long long)(src + off);
            unsigned la = dstbase + off;
            asm volatile("global_load_async_to_lds_b128 %0, %1, off"
                         :: "v"(la), "v"(ga) : "memory");
        }
    };

    float ps[8] = {0,0,0,0,0,0,0,0};
    int rowbase = m0 + hsel * 8;          // row of acc element v is rowbase+v

    stage(0, 0);
    for (int b = 0; b < NBATCH; ++b) {
        if (b + 1 < NBATCH) {
            stage(b + 1, (b + 1) & 1);
            asm volatile("s_wait_asynccnt 0x10" ::: "memory"); // prev 16 done
        } else {
            asm volatile("s_wait_asynccnt 0x0" ::: "memory");
        }
        __syncthreads();                  // panel b visible to all waves

        const _Float16* panel = &buf[b & 1][0];
#pragma unroll
        for (int half = 0; half < 2; ++half) {
            int t  = cb + half * 4;       // column tile within panel (0..7)
            const _Float16* bb = panel + (size_t)(t * 16 + l15) * DDIM + hsel * 8;
            v8f acc = {};
#pragma unroll
            for (int ks = 0; ks < 8; ++ks) {
                v8h lo = *(const v8h*)(bb + ks * 32);
                v8h hi = *(const v8h*)(bb + ks * 32 + 16);
                v16h bfrag = __builtin_shufflevector(lo, hi,
                    0,1,2,3,4,5,6,7,8,9,10,11,12,13,14,15);
                acc = __builtin_amdgcn_wmma_f32_16x16x32_f16(
                    false, a[ks], false, bfrag, (short)0, acc, false, false);
            }
            int col = b * NB + t * 16 + l15;
#pragma unroll
            for (int v = 0; v < 8; ++v) {
                float e = __expf(fmaf(acc[v], INV_T, -MAXS)); // exp(sim - 2)
                if (rowbase + v == col) e = 0.0f;             // mask diagonal
                ps[v] += e;
            }
        }
        __syncthreads();                  // done reading buf[b&1]
    }

    // reduce each row-sum across the 16-lane half (cols differ, rows shared)
#pragma unroll
    for (int v = 0; v < 8; ++v) {
        float s = ps[v];
#pragma unroll
        for (int m = 1; m < 16; m <<= 1) s += __shfl_xor(s, m, 32);
        if (l15 == 0)
            atomicAdd(&rowsum[rt * 16 + hsel * 8 + v], s);
    }
    __syncthreads();

    if (tid < 32) {
        float lse = MAXS + __logf(rowsum[tid]);
        int gr = blockIdx.x * 32 + tid;
        atomicAdd(&out[(gr < NROW) ? 0 : 1], lse * INV_N);
    }
}

// ---------------------------------------------------------------------------
extern "C" void kernel_launch(void* const* d_in, const int* in_sizes, int n_in,
                              void* d_out, int out_size, void* d_ws, size_t ws_size,
                              hipStream_t stream) {
    const float* zi = (const float*)d_in[0];   // (4096, 256) f32
    const float* zj = (const float*)d_in[1];   // (4096, 256) f32
    float* out = (float*)d_out;                // 2 floats: loss_i, loss_j
    _Float16* H = (_Float16*)d_ws;             // 8192 x 256 f16 (4 MB)

    zero_out<<<1, 64, 0, stream>>>(out);
    normalize_f16<<<TWO_N / 8, 256, 0, stream>>>(zi, zj, H);
    pos_subtract<<<NROW / 8, 256, 0, stream>>>(H, out);
    ntxent_gram<<<TWO_N / 32, 256, 0, stream>>>(H, out);
}
